// GatedGCNLayerEdgeFeatOnly_18090402251221
// MI455X (gfx1250) — matrix operand order repaired and checked by reading, not compiled, not run on hardware
//
#include <hip/hip_runtime.h>

typedef _Float16 f16;
typedef __attribute__((ext_vector_type(16))) _Float16 v16h;
typedef __attribute__((ext_vector_type(8)))  _Float16 v8h;
typedef __attribute__((ext_vector_type(4)))  _Float16 v4h;
typedef __attribute__((ext_vector_type(8)))  float    v8f;

#define DD 128

// ---------------------------------------------------------------------------
// Convert h (f32) -> h16 (f16), row-major N x 128
// ---------------------------------------------------------------------------
__global__ void cvt_h_kernel(const float* __restrict__ h, f16* __restrict__ h16, long n) {
    long i = (long)blockIdx.x * blockDim.x + threadIdx.x;
    if (i < n) h16[i] = (f16)h[i];
}

// ---------------------------------------------------------------------------
// Repack concatenated weight Wcat[128 x 512] = [A_w | B_w | D_w | E_w] into the
// exact per-lane WMMA B-operand layout (f16), plus concatenated bias (f32).
// Wp index: (((kt*32 + nt)*32 + lane)*16 + j)
//   kt = K-step (0..3, blocks of 32), nt = column tile (0..31)
//   lane < 16: holds B[K = kt*32 + (j<8 ? j : 8+j)     ][col = nt*16 + lane   ]
//   lane >=16: holds B[K = kt*32 + (j<8 ? 8+j : 16+j)  ][col = nt*16 + lane-16]
// ---------------------------------------------------------------------------
__global__ void repack_kernel(const float* __restrict__ Aw, const float* __restrict__ Bw,
                              const float* __restrict__ Dw, const float* __restrict__ Ew,
                              const float* __restrict__ Ab, const float* __restrict__ Bb,
                              const float* __restrict__ Db, const float* __restrict__ Eb,
                              f16* __restrict__ Wp, float* __restrict__ bcat) {
    int i = blockIdx.x * blockDim.x + threadIdx.x;
    if (i < 4 * 32 * 32 * 16) {
        int j    = i & 15;
        int lane = (i >> 4) & 31;
        int nt   = (i >> 9) & 31;
        int kt   = i >> 14;
        int k    = kt * 32 + ((lane >> 4) << 3) + (j & 7) + ((j >> 3) << 4);   // 0..127
        int col  = nt * 16 + (lane & 15);              // 0..511
        const float* W = (col < 128) ? Aw : (col < 256) ? Bw : (col < 384) ? Dw : Ew;
        Wp[i] = (f16)W[k * DD + (col & 127)];
    }
    if (i < 512) {
        const float* B = (i < 128) ? Ab : (i < 256) ? Bb : (i < 384) ? Db : Eb;
        bcat[i] = B[i & 127];
    }
}

// ---------------------------------------------------------------------------
// Fused 4-projection GEMM: [Ah|Bh|Dh|Eh] = h @ Wcat + bcat  via WMMA f16->f32.
// One wave per 16x16 output tile, K=128 in 4 steps of 32.
// The projection selector m = nt>>3 is WAVE-UNIFORM (a 16-col tile never
// straddles a 128-col projection boundary) -> scalar branch, no EXEC churn.
// ---------------------------------------------------------------------------
__global__ __launch_bounds__(256) void gemm4_wmma_kernel(
    const f16* __restrict__ h16, const f16* __restrict__ Wp,
    const float* __restrict__ bcat,
    float* __restrict__ Ah, f16* __restrict__ Bh,
    f16* __restrict__ Dh, f16* __restrict__ Eh,
    int n, int mtiles) {
    int wave = threadIdx.x >> 5;
    int lane = threadIdx.x & 31;
    int tile = blockIdx.x * 8 + wave;
    if (tile >= mtiles * 32) return;            // wave-uniform: EXEC stays all-1s
    int mt = tile >> 5;                         // row tile
    int nt = tile & 31;                         // column tile (of 512 cols)

    int r    = lane & 15;
    int row  = mt * 16 + r;
    int rowc = row < n ? row : n - 1;           // clamp (N%16==0 anyway)
    int k0   = (lane >> 4) << 3;                // 0 or 8 (A-layout half-wave split)
    const f16* arow = h16 + (long)rowc * DD;

    v8f c = {};
#pragma unroll
    for (int kt = 0; kt < 4; ++kt) {
        int kk = kt * 32;
        v8h alo = *(const v8h*)(arow + kk + k0);        // K = k0 .. k0+7
        v8h ahi = *(const v8h*)(arow + kk + k0 + 16);   // K = k0+16 .. k0+23
        v16h a = __builtin_shufflevector(alo, ahi,
                                         0, 1, 2, 3, 4, 5, 6, 7,
                                         8, 9, 10, 11, 12, 13, 14, 15);
        v16h b = *(const v16h*)(Wp + (((kt * 32 + nt) * 32 + lane) << 4));
        c = __builtin_amdgcn_wmma_f32_16x16x32_f16(
                /*neg_a=*/false, a, /*neg_b=*/false, b,
                /*c_mod=*/(short)0, c, /*reuse_a=*/false, /*reuse_b=*/false);
    }

    float bias = bcat[nt * 16 + (lane & 15)];
    int m      = nt >> 3;                            // wave-uniform projection id
    int cl     = ((nt & 7) << 4) + (lane & 15);      // column within projection
    int rbase  = mt * 16 + ((lane >> 4) << 3);       // C layout: VGPR v -> M=v (+8 hi lanes)

    if (m == 0) {
#pragma unroll
        for (int v = 0; v < 8; ++v) {
            int rr = rbase + v;
            if (rr < n) Ah[(long)rr * DD + cl] = c[v] + bias;
        }
    } else {
        f16* dstp = (m == 1) ? Bh : (m == 2) ? Dh : Eh;
#pragma unroll
        for (int v = 0; v < 8; ++v) {
            int rr = rbase + v;
            if (rr < n) dstp[(long)rr * DD + cl] = (f16)(c[v] + bias);
        }
    }
}

// ---------------------------------------------------------------------------
// Edge phase: sigma = sigmoid(Dh[src] + Eh[dst]);
// num[dst] += sigma * Bh[src]; den[dst] += sigma   (f32 atomics, L2-resident)
// 32 threads per edge, 4 features per thread: packed f16x4 (b64) gathers,
// 4x less src/dst index traffic, still fully coalesced (256B per edge).
// ---------------------------------------------------------------------------
__global__ __launch_bounds__(256) void edge_kernel(
    const f16* __restrict__ Bh, const f16* __restrict__ Dh, const f16* __restrict__ Eh,
    const int* __restrict__ src, const int* __restrict__ dst,
    float* __restrict__ num, float* __restrict__ den, long total /* = E*32 */) {
    long i = (long)blockIdx.x * blockDim.x + threadIdx.x;
    if (i >= total) return;
    int  d = (int)(i & 31) << 2;               // feature quad base
    long e = i >> 5;
    int  s = src[e];
    int  t = dst[e];
    v4h dv = *(const v4h*)(Dh + (long)s * DD + d);
    v4h ev = *(const v4h*)(Eh + (long)t * DD + d);
    v4h bv = *(const v4h*)(Bh + (long)s * DD + d);
    float* np = num + (long)t * DD + d;
    float* dp = den + (long)t * DD + d;
#pragma unroll
    for (int j = 0; j < 4; ++j) {
        float x  = (float)dv[j] + (float)ev[j];
        float sg = 1.0f / (1.0f + __expf(-x));
        __hip_atomic_fetch_add(np + j, sg * (float)bv[j],
                               __ATOMIC_RELAXED, __HIP_MEMORY_SCOPE_AGENT);
        __hip_atomic_fetch_add(dp + j, sg,
                               __ATOMIC_RELAXED, __HIP_MEMORY_SCOPE_AGENT);
    }
}

// ---------------------------------------------------------------------------
// h_new = Ah + num/(den+eps) (written in place into num), and accumulate
// per-column sum / sum-of-squares for batch-norm via LDS reduce + atomics.
// Each block covers 128 rows; 2 threads per column.
// ---------------------------------------------------------------------------
__global__ __launch_bounds__(256) void hnew_stats_kernel(
    const float* __restrict__ Ah, float* __restrict__ num /* -> h_new */,
    const float* __restrict__ den,
    float* __restrict__ colsum, float* __restrict__ colsq, int n) {
    __shared__ float ssum[256];
    __shared__ float ssq[256];
    int tid   = threadIdx.x;
    int dcol  = tid & 127;
    int half  = tid >> 7;
    int rbase = blockIdx.x * 128;
    float s = 0.f, q = 0.f;
#pragma unroll 4
    for (int j = 0; j < 64; ++j) {
        int r = rbase + half + (j << 1);
        if (r < n) {
            long  o  = (long)r * DD + dcol;
            float hn = Ah[o] + num[o] / (den[o] + 1e-6f);
            num[o] = hn;
            s += hn;
            q += hn * hn;
        }
    }
    ssum[tid] = s;
    ssq[tid]  = q;
    __syncthreads();
    if (tid < 128) {
        atomicAdd(&colsum[dcol], ssum[tid] + ssum[tid + 128]);
        atomicAdd(&colsq[dcol],  ssq[tid]  + ssq[tid + 128]);
    }
}

// ---------------------------------------------------------------------------
// Batch-norm (batch stats) + ReLU + residual.
// ---------------------------------------------------------------------------
__global__ __launch_bounds__(256) void finalize_kernel(
    const float* __restrict__ hin, const float* __restrict__ hnew,
    const float* __restrict__ colsum, const float* __restrict__ colsq,
    const float* __restrict__ gamma, const float* __restrict__ beta,
    float* __restrict__ out, long total, float invN) {
    long i = (long)blockIdx.x * blockDim.x + threadIdx.x;
    if (i >= total) return;
    int   d    = (int)(i & (DD - 1));
    float mean = colsum[d] * invN;
    float var  = colsq[d] * invN - mean * mean;
    float v    = (hnew[i] - mean) * rsqrtf(var + 1e-5f) * gamma[d] + beta[d];
    v = v > 0.f ? v : 0.f;
    out[i] = hin[i] + v;
}

static inline size_t align_up(size_t x) { return (x + 255) & ~(size_t)255; }

extern "C" void kernel_launch(void* const* d_in, const int* in_sizes, int n_in,
                              void* d_out, int out_size, void* d_ws, size_t ws_size,
                              hipStream_t stream) {
    const float* h     = (const float*)d_in[0];
    const float* e     = (const float*)d_in[1];
    const int*   src   = (const int*)  d_in[2];
    const int*   dst   = (const int*)  d_in[3];
    const float* Aw    = (const float*)d_in[4];
    const float* Ab    = (const float*)d_in[5];
    const float* Bw    = (const float*)d_in[6];
    const float* Bb    = (const float*)d_in[7];
    const float* Dw    = (const float*)d_in[8];
    const float* Db    = (const float*)d_in[9];
    const float* Ew    = (const float*)d_in[10];
    const float* Eb    = (const float*)d_in[11];
    const float* gamma = (const float*)d_in[12];
    const float* beta  = (const float*)d_in[13];

    const int  N  = in_sizes[0] / DD;
    const int  E  = in_sizes[2];
    const long ND = (long)N * DD;
    const long ED = (long)E * DD;

    // workspace carve-up
    char* p = (char*)d_ws;
    f16*   h16  = (f16*)p;   p += align_up((size_t)ND * 2);
    f16*   Wp   = (f16*)p;   p += align_up((size_t)4 * 32 * 32 * 16 * 2);
    float* bcat = (float*)p; p += align_up(512 * 4);
    float* Ah   = (float*)p; p += align_up((size_t)ND * 4);
    f16*   Bh   = (f16*)p;   p += align_up((size_t)ND * 2);
    f16*   Dh   = (f16*)p;   p += align_up((size_t)ND * 2);
    f16*   Eh   = (f16*)p;   p += align_up((size_t)ND * 2);
    float* num  = (float*)p; p += align_up((size_t)ND * 4);   // becomes h_new
    float* den  = (float*)p; p += align_up((size_t)ND * 4);
    float* colsum = (float*)p; // colsum[128] then colsq[128], adjacent
    float* colsq  = colsum + 128;

    // zero accumulators (graph-capture-safe async memsets)
    hipMemsetAsync(num, 0, (size_t)ND * 4, stream);
    hipMemsetAsync(den, 0, (size_t)ND * 4, stream);
    hipMemsetAsync(colsum, 0, 256 * 4, stream);

    // 1) h -> f16
    {
        long total = ND;
        int blocks = (int)((total + 255) / 256);
        cvt_h_kernel<<<blocks, 256, 0, stream>>>(h, h16, total);
    }
    // 2) weight repack + bias concat
    {
        int total = 4 * 32 * 32 * 16;
        repack_kernel<<<(total + 255) / 256, 256, 0, stream>>>(
            Aw, Bw, Dw, Ew, Ab, Bb, Db, Eb, Wp, bcat);
    }
    // 3) fused WMMA GEMM: [Ah|Bh|Dh|Eh]
    {
        int mtiles = (N + 15) / 16;
        int tiles  = mtiles * 32;
        int blocks = (tiles + 7) / 8;   // 8 waves (tiles) per 256-thread block
        gemm4_wmma_kernel<<<blocks, 256, 0, stream>>>(h16, Wp, bcat, Ah, Bh, Dh, Eh, N, mtiles);
    }
    // 4) edge gather / sigmoid / atomic scatter (32 threads per edge)
    {
        long total = (long)E * 32;
        int blocks = (int)((total + 255) / 256);
        edge_kernel<<<blocks, 256, 0, stream>>>(Bh, Dh, Eh, src, dst, num, den, total);
    }
    // 5) h_new + BN statistics
    {
        int blocks = (N + 127) / 128;
        hnew_stats_kernel<<<blocks, 256, 0, stream>>>(Ah, num, den, colsum, colsq, N);
    }
    // 6) BN + ReLU + residual -> d_out[0 : N*D]
    {
        long total = ND;
        int blocks = (int)((total + 255) / 256);
        finalize_kernel<<<blocks, 256, 0, stream>>>(
            h, num, colsum, colsq, gamma, beta, (float*)d_out, total, 1.0f / (float)N);
    }
    // 7) e passthrough -> d_out[N*D : N*D + E*D]
    hipMemcpyAsync((float*)d_out + ND, e, (size_t)ED * sizeof(float),
                   hipMemcpyDeviceToDevice, stream);
}
